// CPCModel_50878182588587
// MI455X (gfx1250) — matrix operand order repaired and checked by reading, not compiled, not run on hardware
//
#include <hip/hip_runtime.h>

// ---------------------------------------------------------------------------
// Fused FCM-push pipeline for MI455X (gfx1250, wave32, WMMA).
//   GEMM1: cross = V . P^T     -> w  = 1/max(|v|^2+|c|^2-2cross, eps^2)
//   GEMM2: target = belong . P -> x  = v + 0.5*(target - v)
//   GEMM3: cross2 = X . P^T    -> w2 -> out = w2 / sum_k w2   (final clip == id)
// One workgroup (128 thr = 4 waves) fuses everything for a 64-row tile in
// ~298KB of the 320KB WGP LDS. Proto chunks staged via the Tensor Data Mover
// (tensor_load_to_lds + s_wait_tensorcnt), then converted ONCE per chunk into
// bf16 copies (row-major + transposed) so every WMMA fragment is exactly two
// contiguous ds_load_b128s. Reciprocals via v_rcp_f32 (no IEEE div expansion).
// ---------------------------------------------------------------------------

typedef __bf16 bf16;
typedef bf16         v16bf __attribute__((ext_vector_type(16)));
typedef bf16         v8bf  __attribute__((ext_vector_type(8)));
typedef float        v8f   __attribute__((ext_vector_type(8)));
typedef unsigned int v4u   __attribute__((ext_vector_type(4)));
typedef int          v8i   __attribute__((ext_vector_type(8)));
typedef int          v4i   __attribute__((ext_vector_type(4)));

#define D_DIM   256
#define K_DIM   512
#define ROWS_WG 64
#define THREADS 128
#define CHUNK   32        // protos per LDS staging chunk
#define NCHUNK  (K_DIM / CHUNK)

// LDS byte offsets (total 298,240 B < 320KB WGP LDS)
#define OFF_P    0        // f32  [32][256] TDM landing buffer  =  32768
#define OFF_PB   32768    // bf16 [32][256] row-major copy      =  16384
#define OFF_PBT  49152    // bf16 [256][32] transposed copy     =  16384
#define OFF_W    65536    // f32  [64][512] weights w / w2      = 131072
#define OFF_VB   196608   // bf16 [64][256] V tile; aliased by X tile = 32768
#define OFF_BEL  229376   // bf16 [64][512] belong              =  65536
#define OFF_C2   294912   // f32  [512] proto sq norms          =   2048
#define OFF_V2   296960   // f32  [64]
#define OFF_X2   297216   // f32  [64]
#define OFF_SUM  297472   // f32  [64]
#define OFF_TMP  297728   // f32  [128]
#define LDS_TOTAL 298240

union PairU { unsigned u; bf16 h[2]; };

#define HAVE_TDM __has_builtin(__builtin_amdgcn_tensor_load_to_lds)

// Fast reciprocal: v_rcp_f32 (~1 ulp) instead of the IEEE divide expansion.
__device__ __forceinline__ float fast_rcp(float x)
{
#if __has_builtin(__builtin_amdgcn_rcpf)
  return __builtin_amdgcn_rcpf(x);
#else
  return 1.0f / x;
#endif
}

// Stage one 32x256 f32 proto chunk (contiguous 32KB) into LDS via TDM, then
// convert once into the requested bf16 copies (amortized over all 4 waves).
__device__ __forceinline__ void stage_protos(const float* __restrict__ gP,
                                             int chunk, float* sP, bf16* sPb,
                                             bf16* sPbT, bool buildRow, bool buildT)
{
  __syncthreads();                       // previous consumers of P buffers done
#if HAVE_TDM
  if (threadIdx.x < 32) {                // wave 0 issues the TDM descriptor
    unsigned long long ga = (unsigned long long)(const void*)(gP + (size_t)chunk * CHUNK * D_DIM);
    unsigned lds = (unsigned)(size_t)(void*)sP;   // low 32 bits = LDS offset
    v4u g0;
    g0[0] = 1u;                                   // count = 1 (valid D#)
    g0[1] = lds;                                  // lds_addr
    g0[2] = (unsigned)(ga & 0xffffffffu);         // global_addr[31:0]
    g0[3] = (unsigned)((ga >> 32) & 0x01ffffffu)  // global_addr[56:32]
            | (2u << 30);                         // type = 2 ("image")
    v8i g1;
    g1[0] = (int)(2u << 16);                      // data_size = 2 -> 4B elems
    g1[1] = (int)((unsigned)D_DIM << 16);         // tensor_dim0 = 256 (lo16)
    g1[2] = (int)((unsigned)K_DIM << 16);         // dim0 hi=0 ; tensor_dim1 = 512
    g1[3] = (int)((unsigned)D_DIM << 16);         // dim1 hi=0 ; tile_dim0 = 256
    g1[4] = (int)CHUNK;                           // tile_dim1 = 32 ; tile_dim2 = 0
    g1[5] = D_DIM;                                // tensor_dim0_stride = 256
    g1[6] = 0;
    g1[7] = 0;
    v4i g2 = {0, 0, 0, 0};                        // 2-D tile: groups 2/3 unused
    v4i g3 = {0, 0, 0, 0};
    v8i g4 = {0, 0, 0, 0, 0, 0, 0, 0};            // extra arg of 6-arg builtin
    __builtin_amdgcn_tensor_load_to_lds(g0, g1, g2, g3, g4, 0);
    __builtin_amdgcn_s_wait_tensorcnt(0);
  }
#else
  for (int i = (int)threadIdx.x * 4; i < CHUNK * D_DIM; i += THREADS * 4) {
    float4 f = *(const float4*)(gP + (size_t)chunk * CHUNK * D_DIM + i);
    *(float4*)(sP + i) = f;
  }
#endif
  __syncthreads();                       // f32 chunk visible to all waves
  // one-shot f32 -> bf16 conversion (row-major and/or transposed)
  for (int idx = (int)threadIdx.x * 2; idx < CHUNK * D_DIM; idx += THREADS * 2) {
    float a0 = sP[idx], a1 = sP[idx + 1];
    PairU u; u.h[0] = (bf16)a0; u.h[1] = (bf16)a1;
    if (buildRow) *(unsigned*)(sPb + idx) = u.u;
    if (buildT) {
      int p = idx >> 8, d = idx & 255;   // [proto][d] -> [d][proto]
      sPbT[d * CHUNK + p]       = u.h[0];
      sPbT[(d + 1) * CHUNK + p] = u.h[1];
    }
  }
  __syncthreads();                       // bf16 copies ready
}

// A fragment (16x32 bf16): lane (m=l%16, hi=l/16) holds K-values
// [16*hi, 16*hi+16) -> two contiguous 16B runs.
__device__ __forceinline__ v16bf frag_a_bf16(const bf16* s, int stride, int m,
                                             int dbase, int hi)
{
  const bf16* base = s + m * stride + dbase + 16 * hi;
  v8bf r0 = *(const v8bf*)(base);        // ds_load_b128
  v8bf r1 = *(const v8bf*)(base + 8);    // ds_load_b128
  return __builtin_shufflevector(r0, r1, 0, 1, 2, 3, 4, 5, 6, 7,
                                 8, 9, 10, 11, 12, 13, 14, 15);
}

// B fragment (32x16 bf16): lane (n=l%16, hi) holds K = 16*hi + [0,16).
// GEMM1/3: B[k][n] = P[protoRow(n)][dbase+k] -> contiguous in row-major copy.
__device__ __forceinline__ v16bf frag_b_rowK(const bf16* sPb, int protoRow,
                                             int dbase, int hi)
{
  const bf16* base = sPb + protoRow * D_DIM + dbase + 16 * hi;
  v8bf r0 = *(const v8bf*)(base);
  v8bf r1 = *(const v8bf*)(base + 8);
  return __builtin_shufflevector(r0, r1, 0, 1, 2, 3, 4, 5, 6, 7,
                                 8, 9, 10, 11, 12, 13, 14, 15);
}

// GEMM2: B[kk][n] = P[kk][dcol(n)] = P^T[dcol][kk] -> contiguous in transposed
// copy ([256][32] bf16), kk = 16*hi + [0,16).
__device__ __forceinline__ v16bf frag_b_colK(const bf16* sPbT, int dcol, int hi)
{
  const bf16* base = sPbT + dcol * CHUNK + 16 * hi;
  v8bf r0 = *(const v8bf*)(base);
  v8bf r1 = *(const v8bf*)(base + 8);
  return __builtin_shufflevector(r0, r1, 0, 1, 2, 3, 4, 5, 6, 7,
                                 8, 9, 10, 11, 12, 13, 14, 15);
}

// One proto-chunk of distance->weight for a wave's 16 rows (GEMM1 / GEMM3).
__device__ __forceinline__ void dist_weights_pass(const bf16* sA, const float* sN2,
                                                  const float* sC2, float* sW,
                                                  const bf16* sPb, int chunk,
                                                  int rowBase, int lo, int hi)
{
  v8f acc[2] = {};
#pragma unroll
  for (int kd = 0; kd < 8; ++kd) {
    v16bf a = frag_a_bf16(sA, D_DIM, rowBase + lo, kd * 32, hi);
#pragma unroll
    for (int pt = 0; pt < 2; ++pt) {
      v16bf b = frag_b_rowK(sPb, pt * 16 + lo, kd * 32, hi);
      acc[pt] = __builtin_amdgcn_wmma_f32_16x16x32_bf16(
          false, a, false, b, (short)0, acc[pt], false, false);
    }
  }
  // C layout: VGPR r -> row (r + 8*hi), lane -> col lo.
#pragma unroll
  for (int pt = 0; pt < 2; ++pt) {
    int proto = chunk * CHUNK + pt * 16 + lo;
    float c2v = sC2[proto];
#pragma unroll
    for (int r = 0; r < 8; ++r) {
      int row = rowBase + r + 8 * hi;
      float sqd = fmaxf(sN2[row] + c2v - 2.0f * acc[pt][r], 0.0f);
      // w = clamp(sqrt(sqd), EPS)^-2  ==  1 / max(sqd, EPS^2)
      sW[row * K_DIM + proto] = fast_rcp(fmaxf(sqd, 1e-16f));
    }
  }
}

// Sum 512 f32 per row (64 rows, 2 threads/row) into sDst[64].
__device__ __forceinline__ void rowsum_512(const float* sWsrc, float* sDst, float* sTmp)
{
  int row = threadIdx.x >> 1, part = threadIdx.x & 1;
  float s = 0.f;
  const float* p = sWsrc + row * K_DIM + part * 256;
  for (int j = 0; j < 256; ++j) s += p[j];
  sTmp[threadIdx.x] = s;
  __syncthreads();
  if (threadIdx.x < ROWS_WG)
    sDst[threadIdx.x] = sTmp[2 * threadIdx.x] + sTmp[2 * threadIdx.x + 1];
  __syncthreads();
}

__global__ __launch_bounds__(THREADS, 1)
void fcm_push_wmma_kernel(const float* __restrict__ gV,
                          const float* __restrict__ gP,
                          float* __restrict__ gOut)
{
  extern __shared__ char smem[];
  float* sP   = (float*)(smem + OFF_P);
  bf16*  sPb  = (bf16*)(smem + OFF_PB);
  bf16*  sPbT = (bf16*)(smem + OFF_PBT);
  float* sW   = (float*)(smem + OFF_W);
  bf16*  sVb  = (bf16*)(smem + OFF_VB);
  bf16*  sX   = (bf16*)(smem + OFF_VB);   // aliases sVb (dead after phase 1)
  bf16*  sBel = (bf16*)(smem + OFF_BEL);
  float* sC2  = (float*)(smem + OFF_C2);
  float* sV2  = (float*)(smem + OFF_V2);
  float* sX2  = (float*)(smem + OFF_X2);
  float* sSum = (float*)(smem + OFF_SUM);
  float* sTmp = (float*)(smem + OFF_TMP);

  const int tid  = threadIdx.x;
  const int lane = tid & 31;
  const int wid  = tid >> 5;            // wave id 0..3 (wave32)
  const int lo   = lane & 15;
  const int hi   = lane >> 4;
  const int rowBase = wid * 16;         // wave's 16 rows inside the tile
  const long wgBase = (long)blockIdx.x * ROWS_WG;

  // ---- Phase 0: load V tile -> bf16 LDS copy + |v|^2 -----------------------
  {
    int row = tid >> 1, half = tid & 1;
    const float* src = gV + (wgBase + row) * D_DIM + half * 128;
    bf16* dst = sVb + row * D_DIM + half * 128;
    float s = 0.f;
#pragma unroll 8
    for (int i = 0; i < 128; i += 4) {
      float4 f = *(const float4*)(src + i);
      s += f.x * f.x + f.y * f.y + f.z * f.z + f.w * f.w;
      PairU u0; u0.h[0] = (bf16)f.x; u0.h[1] = (bf16)f.y;
      PairU u1; u1.h[0] = (bf16)f.z; u1.h[1] = (bf16)f.w;
      *(unsigned*)(dst + i)     = u0.u;
      *(unsigned*)(dst + i + 2) = u1.u;
    }
    sTmp[tid] = s;
    __syncthreads();
    if (tid < ROWS_WG) sV2[tid] = sTmp[2 * tid] + sTmp[2 * tid + 1];
    // stage_protos() starts with a barrier -> sV2/sVb visible before use
  }

  // ---- Phase 1: GEMM1 (V.P^T) -> raw weights w into sW ---------------------
  for (int c = 0; c < NCHUNK; ++c) {
    stage_protos(gP, c, sP, sPb, sPbT, true, false);
    if (tid < CHUNK) {                  // proto squared norms (kept for phase 3)
      float s = 0.f;
      const float* p = sP + tid * D_DIM;
      for (int j = 0; j < D_DIM; ++j) s += p[j] * p[j];
      sC2[c * CHUNK + tid] = s;
    }
    __syncthreads();
    dist_weights_pass(sVb, sV2, sC2, sW, sPb, c, rowBase, lo, hi);
  }
  __syncthreads();

  // ---- normalize: belong = w / sum_k w  (bf16, feeds GEMM2) ---------------
  rowsum_512(sW, sSum, sTmp);
  for (int idx = tid * 2; idx < ROWS_WG * K_DIM; idx += THREADS * 2) {
    int row = idx >> 9;
    float inv = fast_rcp(sSum[row]);
    PairU u; u.h[0] = (bf16)(sW[idx] * inv); u.h[1] = (bf16)(sW[idx + 1] * inv);
    *(unsigned*)(sBel + idx) = u.u;
  }
  __syncthreads();

  // ---- Phase 2: GEMM2 target = belong.P ; x = v + 0.5*(target - v) --------
  for (int g = 0; g < 4; ++g) {         // 64 D-columns per group
    v8f acc[4] = {};
    for (int c = 0; c < NCHUNK; ++c) {
      stage_protos(gP, c, sP, sPb, sPbT, false, true);
      // single K=32 step per 32-proto chunk
      v16bf a = frag_a_bf16(sBel, K_DIM, rowBase + lo, c * CHUNK, hi);
#pragma unroll
      for (int dt = 0; dt < 4; ++dt) {
        v16bf b = frag_b_colK(sPbT, g * 64 + dt * 16 + lo, hi);
        acc[dt] = __builtin_amdgcn_wmma_f32_16x16x32_bf16(
            false, a, false, b, (short)0, acc[dt], false, false);
      }
    }
#pragma unroll
    for (int dt = 0; dt < 4; ++dt) {
      int col = g * 64 + dt * 16 + lo;
#pragma unroll
      for (int r = 0; r < 8; ++r) {
        int row = rowBase + r + 8 * hi;
        float v = gV[(wgBase + row) * D_DIM + col];   // exact f32 v (L2-hot)
        float x = v + 0.5f * (acc[dt][r] - v);        // PUSH_DEG = 0.5
        sX[row * D_DIM + col] = (bf16)x;
      }
    }
  }
  __syncthreads();

  // ---- |x|^2 ---------------------------------------------------------------
  {
    int row = tid >> 1, half = tid & 1;
    const bf16* p = sX + row * D_DIM + half * 128;
    float s = 0.f;
    for (int i = 0; i < 128; ++i) { float f = (float)p[i]; s += f * f; }
    sTmp[tid] = s;
    __syncthreads();
    if (tid < ROWS_WG) sX2[tid] = sTmp[2 * tid] + sTmp[2 * tid + 1];
  }

  // ---- Phase 3: GEMM3 (X.P^T) -> w2 into sW --------------------------------
  for (int c = 0; c < NCHUNK; ++c) {
    stage_protos(gP, c, sP, sPb, sPbT, true, false);  // barrier covers sX2 too
    dist_weights_pass(sX, sX2, sC2, sW, sPb, c, rowBase, lo, hi);
  }
  __syncthreads();

  // ---- normalize + write: out = w2 / sum_k w2.
  // (The reference's final divide by clip(sum enc,1) is identity: the
  //  normalized memberships sum to exactly 1 with this same normalizer.)
  rowsum_512(sW, sSum, sTmp);
  float* outBase = gOut + wgBase * K_DIM;
  for (int idx = tid * 4; idx < ROWS_WG * K_DIM; idx += THREADS * 4) {
    int row = idx >> 9;
    float inv = fast_rcp(sSum[row]);
    float4 wv = *(const float4*)(sW + idx);
    float4 o;
    o.x = wv.x * inv; o.y = wv.y * inv; o.z = wv.z * inv; o.w = wv.w * inv;
    *(float4*)(outBase + idx) = o;      // coalesced 128b stores
  }
}

extern "C" void kernel_launch(void* const* d_in, const int* in_sizes, int n_in,
                              void* d_out, int out_size, void* d_ws, size_t ws_size,
                              hipStream_t stream)
{
  (void)n_in; (void)out_size; (void)d_ws; (void)ws_size;
  const float* gV = (const float*)d_in[0];   // encodedData [B,N,D] f32
  const float* gP = (const float*)d_in[1];   // protos      [K,D]   f32
  float* gOut = (float*)d_out;               // [B,N,K] f32
  int rows = in_sizes[0] / D_DIM;            // B*N = 65536
  int blocks = rows / ROWS_WG;               // 1024 workgroups
  fcm_push_wmma_kernel<<<dim3(blocks), dim3(THREADS), LDS_TOTAL, stream>>>(gV, gP, gOut);
}
